// MoELayer_59846074302684
// MI455X (gfx1250) — matrix operand order repaired and checked by reading, not compiled, not run on hardware
//
#include <hip/hip_runtime.h>
#include <hip/hip_bf16.h>
#include <math.h>

// Problem constants (from reference)
#define NTOK   8192      // B*M*H*W
#define CIN    512
#define HID    1024
#define OUTD   512
#define NEXP   8
#define TOPK   4

typedef __attribute__((ext_vector_type(16))) __bf16 v16bf;
typedef __attribute__((ext_vector_type(8)))  float  v8f;

union Frag {
    uint32_t u[8];
    v16bf    bf;
};

// Round-to-nearest-even fp32 -> bf16, pack two into one dword (lo = a, hi = b).
static __device__ __forceinline__ uint32_t pk_bf16(float a, float b) {
    union { float f; uint32_t u; } ua, ub;
    ua.f = a; ub.f = b;
    uint32_t lo = (ua.u + 0x7FFFu + ((ua.u >> 16) & 1u)) >> 16;
    uint32_t hi = (ub.u + 0x7FFFu + ((ub.u >> 16) & 1u)) >> 16;
    return (lo & 0xFFFFu) | (hi << 16);
}

static __device__ __forceinline__ float gelu_exact(float x) {
    return 0.5f * x * (1.0f + erff(x * 0.70710678118654752440f));
}

// ---------------------------------------------------------------------------
// Kernel 1: gating + sigmoid + top-4 (first-index tie-break) + prob normalize
// ---------------------------------------------------------------------------
__global__ __launch_bounds__(256)
void gate_topk_kernel(const float* __restrict__ x,
                      const float* __restrict__ gw,   // [C, E]
                      const float* __restrict__ gb,   // [E]
                      const float* __restrict__ eb,   // [E]
                      int*   __restrict__ tk_idx,     // [N, 4]
                      float* __restrict__ tk_p) {     // [N, 4]
    int n = blockIdx.x * blockDim.x + threadIdx.x;
    if (n >= NTOK) return;
    const float* xr = x + (size_t)n * CIN;

    float g[NEXP];
#pragma unroll
    for (int e = 0; e < NEXP; ++e) g[e] = 0.0f;
    for (int c = 0; c < CIN; ++c) {
        float xv = xr[c];
#pragma unroll
        for (int e = 0; e < NEXP; ++e) g[e] += xv * gw[c * NEXP + e];
    }

    float prob[NEXP], logit[NEXP];
#pragma unroll
    for (int e = 0; e < NEXP; ++e) {
        float go = g[e] + gb[e];
        prob[e]  = 1.0f / (1.0f + expf(-go));
        logit[e] = go + eb[e];
    }

    int  sel[TOPK];
    bool used[NEXP];
#pragma unroll
    for (int e = 0; e < NEXP; ++e) used[e] = false;
#pragma unroll
    for (int i = 0; i < TOPK; ++i) {
        float best = -INFINITY; int bi = 0;
#pragma unroll
        for (int e = 0; e < NEXP; ++e) {
            if (!used[e] && logit[e] > best) { best = logit[e]; bi = e; }
        }
        used[bi] = true;
        sel[i] = bi;
    }
    float ps = 0.0f;
#pragma unroll
    for (int i = 0; i < TOPK; ++i) ps += prob[sel[i]];
    float inv = 1.0f / ps;
#pragma unroll
    for (int i = 0; i < TOPK; ++i) {
        tk_idx[n * TOPK + i] = sel[i];
        tk_p[n * TOPK + i]   = prob[sel[i]] * inv;
    }
}

// ---------------------------------------------------------------------------
// Kernel 2: per-expert GEMM1 + GELU fused with reduced GEMM2 (4 live columns)
//   grid = (64 row tiles, 8 experts), block = 256 (8 wave32)
//   Each wave owns a 16-row block; its A operand (16x512 bf16 = 128 VGPRs)
//   is register-resident for the whole kernel. Only the w1 column slice is
//   staged through LDS per iteration, so each v_wmma_f32_16x16x32_bf16
//   costs just 2 ds_load_b128 (B fragment).
// ---------------------------------------------------------------------------
__global__ __launch_bounds__(256)
void expert_kernel(const float* __restrict__ x,     // [N, C]
                   const float* __restrict__ w1,    // [E, C, HID]
                   const float* __restrict__ b1,    // [E, HID]
                   const float* __restrict__ w2,    // [E, HID, OUT]
                   const float* __restrict__ b2,    // [E, OUT]
                   float* __restrict__ eo4) {       // [E, N, 4]
    __shared__ uint32_t Bsl[16 * 32 * 8];       //  16 KB: [ks][lane][8 dw]
    __shared__ float    AccL[128 * 4];          //   2 KB

    const int e    = blockIdx.y;
    const int row0 = blockIdx.x * 128;
    const int tid  = threadIdx.x;
    const int wave = tid >> 5;
    const int lane = tid & 31;
    const int nlo  = lane & 15;   // N column within tile / M row within block
    const int hih  = lane >> 4;

    // ---- A fragments for this wave's 16 rows, all K=512: register-resident.
    //      A 16-bit fragment layout: dword d -> K0 = (d<4 ? 2d : 16+2(d-4)) + hi*8
    Frag af[16];
    {
        const int row = row0 + wave * 16 + nlo;
        const float* xr = x + (size_t)row * CIN;
#pragma unroll
        for (int ks = 0; ks < 16; ++ks) {
#pragma unroll
            for (int d = 0; d < 8; ++d) {
                const int K0 = ((d < 4) ? (2 * d) : (16 + 2 * (d - 4))) + hih * 8;
                const float2 xv = *(const float2*)(xr + ks * 32 + K0);
                af[ks].u[d] = pk_bf16(xv.x, xv.y);
            }
        }
    }
    for (int i = tid; i < 128 * 4; i += 256) AccL[i] = 0.0f;

    float pacc[8][4];
#pragma unroll
    for (int v = 0; v < 8; ++v)
#pragma unroll
        for (int o = 0; o < 4; ++o) pacc[v][o] = 0.0f;

    const float* w1e = w1 + (size_t)e * CIN * HID;

    for (int cb = 0; cb < HID / 16; ++cb) {
        __syncthreads();   // previous compute done before restaging B
        // ---- stage B slice w1[e][:, cb*16 .. +16] in B-fragment layout ----
        for (int i = tid; i < 16 * 32 * 8; i += 256) {
            int d  = i & 7;
            int l  = (i >> 3) & 31;
            int ks = (i >> 8) & 15;
            int nn = l & 15;
            int hk = l >> 4;
            int k  = ks * 32 + ((d < 4) ? (2 * d) : (16 + 2 * (d - 4))) + hk * 8;
            const float* p = w1e + (size_t)k * HID + cb * 16 + nn;
            Bsl[i] = pk_bf16(p[0], p[HID]);
        }
        __syncthreads();

        // ---- 16x16 hid tile for this wave's row block (rb = wave) ----
        v8f acc = {};
#pragma unroll
        for (int ks = 0; ks < 16; ++ks) {
            Frag fb;
            const uint32_t* bp = &Bsl[(ks * 32 + lane) * 8];
#pragma unroll
            for (int d = 0; d < 8; ++d) fb.u[d] = bp[d];
            acc = __builtin_amdgcn_wmma_f32_16x16x32_bf16(
                false, af[ks].bf, false, fb.bf, (short)0, acc, false, false);
        }

        // ---- bias + exact GELU + contract against 4 live w2 columns ----
        const int   f    = cb * 16 + nlo;                  // hid feature index
        const float bias = b1[e * HID + f];
        const float4 w4  = *(const float4*)(w2 + ((size_t)e * HID + f) * OUTD);
#pragma unroll
        for (int v = 0; v < 8; ++v) {
            float h = gelu_exact(acc[v] + bias);
            pacc[v][0] += h * w4.x;
            pacc[v][1] += h * w4.y;
            pacc[v][2] += h * w4.z;
            pacc[v][3] += h * w4.w;
        }
    }

    // ---- reduce the 16 N-lanes per row via LDS atomics ----
    __syncthreads();
#pragma unroll
    for (int v = 0; v < 8; ++v) {
        int r = wave * 16 + v + 8 * hih;
#pragma unroll
        for (int o = 0; o < 4; ++o)
            atomicAdd(&AccL[r * 4 + o], pacc[v][o]);
    }
    __syncthreads();

    if (tid < 128) {
        float4 r;
        r.x = AccL[tid * 4 + 0] + b2[e * OUTD + 0];
        r.y = AccL[tid * 4 + 1] + b2[e * OUTD + 1];
        r.z = AccL[tid * 4 + 2] + b2[e * OUTD + 2];
        r.w = AccL[tid * 4 + 3] + b2[e * OUTD + 3];
        *(float4*)(eo4 + ((size_t)e * NTOK + row0 + tid) * 4) = r;
    }
}

// ---------------------------------------------------------------------------
// Kernel 3: gather selected experts' channel-i outputs, weighted sum,
//           broadcast scalar across OUT=512.  grid = NTOK blocks of 128.
// ---------------------------------------------------------------------------
__global__ __launch_bounds__(128)
void combine_kernel(const float* __restrict__ eo4,
                    const int*   __restrict__ tk_idx,
                    const float* __restrict__ tk_p,
                    float* __restrict__ out) {
    __shared__ float cmb;
    int n = blockIdx.x;
    if (threadIdx.x == 0) {
        float s = 0.0f;
#pragma unroll
        for (int i = 0; i < TOPK; ++i) {
            int ex = tk_idx[n * TOPK + i];
            s += tk_p[n * TOPK + i] * eo4[((size_t)ex * NTOK + n) * 4 + i];
        }
        cmb = s;
    }
    __syncthreads();
    float4 v; v.x = cmb; v.y = cmb; v.z = cmb; v.w = cmb;
    ((float4*)(out + (size_t)n * OUTD))[threadIdx.x] = v;
}

// ---------------------------------------------------------------------------
extern "C" void kernel_launch(void* const* d_in, const int* in_sizes, int n_in,
                              void* d_out, int out_size, void* d_ws, size_t ws_size,
                              hipStream_t stream) {
    const float* x   = (const float*)d_in[0];
    const float* gw  = (const float*)d_in[1];
    const float* gb  = (const float*)d_in[2];
    const float* eb  = (const float*)d_in[3];
    const float* w1  = (const float*)d_in[4];
    const float* b1  = (const float*)d_in[5];
    const float* w2  = (const float*)d_in[6];
    const float* b2  = (const float*)d_in[7];
    float* out = (float*)d_out;

    char* ws = (char*)d_ws;
    int*   tk_idx = (int*)ws;                                        // 128 KB
    float* tk_p   = (float*)(ws + (size_t)NTOK * TOPK * 4);          // 128 KB
    float* eo4    = (float*)(ws + (size_t)2 * NTOK * TOPK * 4);      //   1 MB

    gate_topk_kernel<<<NTOK / 256, 256, 0, stream>>>(x, gw, gb, eb, tk_idx, tk_p);
    expert_kernel<<<dim3(NTOK / 128, NEXP), 256, 0, stream>>>(x, w1, b1, w2, b2, eo4);
    combine_kernel<<<NTOK, 128, 0, stream>>>(eo4, tk_idx, tk_p, out);
}
